// MultiHeadAttn_9242769622561
// MI455X (gfx1250) — matrix operand order repaired and checked by reading, not compiled, not run on hardware
//
#include <hip/hip_runtime.h>

#define N_HEADS 16
#define D_HEAD  64
#define D_MODEL 1024
#define SEQ     2048
#define BSZ     4
#define LN_EPS  1e-5f
#define QK_SCALE 0.125f   // 1/sqrt(64)

typedef __attribute__((ext_vector_type(16))) _Float16     v16h;
typedef __attribute__((ext_vector_type(8)))  float        v8f;
typedef __attribute__((ext_vector_type(4)))  unsigned int u32x4;

union Frag {
    v16h         h;
    u32x4        q[2];
    unsigned int u[8];
};

static __device__ __forceinline__ v8f wmma_f16(const Frag& a, const Frag& b, v8f c) {
    // D = A(16x32 f16) * B(32x16 f16) + C(16x16 f32)
    return __builtin_amdgcn_wmma_f32_16x16x32_f16(
        false, a.h, false, b.h, (short)0, c, false, false);
}

// ---- CDNA5 async global->LDS copy (tracked by ASYNCcnt, CDNA5 ISA) ----
static __device__ __forceinline__ unsigned lds_addr_of(const void* p) {
    return (unsigned)(size_t)p;   // generic shared ptr: low 32 bits = LDS offset
}
static __device__ __forceinline__ void async_g2l_b128(unsigned lds_addr, const void* gaddr) {
    asm volatile("global_load_async_to_lds_b128 %0, %1, off"
                 :: "v"(lds_addr), "v"(gaddr) : "memory");
}
static __device__ __forceinline__ void wait_async0() {
    asm volatile("s_wait_asynccnt 0x0" ::: "memory");
}

// ---------------------------------------------------------------- converts
__global__ __launch_bounds__(256) void k_f32_to_f16(const float* __restrict__ in,
                                                    _Float16* __restrict__ out) {
    int i = blockIdx.x * 256 + threadIdx.x;     // exact grid, no tail
    out[i] = (_Float16)in[i];
}

// in[K][N] (f32) -> out[N][K] (f16); N = 1<<nshift
__global__ __launch_bounds__(256) void k_w_transpose_f16(const float* __restrict__ in,
                                                         _Float16* __restrict__ out,
                                                         int K, int nshift) {
    int i = blockIdx.x * 256 + threadIdx.x;     // exact grid over K*N
    int k = i >> nshift;
    int n = i & ((1 << nshift) - 1);
    out[(size_t)n * K + k] = (_Float16)in[i];
}

// ---------------------------------------------------------------- GEMM
// C[M,N] = A[M,K] * Bt[N,K]^T, f16 in, f32 accum. Block 128x128, BK=32,
// 8 waves in 4(M)x2(N), each wave 32x64 = 2x4 WMMA tiles.
// Both operand tiles are 128 rows x 32 f16 (64B rows) -> async b128 staging,
// double-buffered so the copy of tile t+1 overlaps the WMMAs of tile t.
#define BM 128
#define BN 128
#define BK 32
#define LDT 40   // f16 elems; 80B rows keep every b128 access 16B-aligned

static __device__ __forceinline__ void stage128x32(_Float16* lds, const _Float16* g,
                                                   int ldg, int tid) {
    const int chunk = tid & 3;   // 16B chunk within the 64B row
    const int r0 = tid >> 2;     // 0..63
#pragma unroll
    for (int it = 0; it < 2; it++) {
        int row = r0 + it * 64;
        async_g2l_b128(lds_addr_of(lds + row * LDT) + chunk * 16,
                       g + (size_t)row * ldg + chunk * 8);
    }
}

__global__ __launch_bounds__(256) void k_gemm_f16(
    const _Float16* __restrict__ A,    // [M][K]
    const _Float16* __restrict__ Bt,   // [N][K]  (pre-transposed weights)
    float* __restrict__ Cf, _Float16* __restrict__ Ch,
    int M, int N, int K, int store_half)
{
    __shared__ __align__(16) _Float16 As[2][BM * LDT];
    __shared__ __align__(16) _Float16 Bs[2][BN * LDT];

    const int tid  = threadIdx.x;
    const int wave = tid >> 5, lane = tid & 31;
    const int wm = wave >> 1, wn = wave & 1;
    const int laneN = lane & 15, half = lane >> 4;
    const int m0 = blockIdx.y * BM, n0 = blockIdx.x * BN;

    const _Float16* Ab = A  + (size_t)m0 * K;
    const _Float16* Bb = Bt + (size_t)n0 * K;

    v8f zero = {};
    v8f acc[2][4];
    for (int i = 0; i < 2; i++)
        for (int j = 0; j < 4; j++) acc[i][j] = zero;

    stage128x32(As[0], Ab, K, tid);
    stage128x32(Bs[0], Bb, K, tid);

    const int NT = K / BK;
    for (int kt = 0; kt < NT; kt++) {
        const int cur = kt & 1;
        wait_async0();        // our async fills of buf[cur] are done
        __syncthreads();      // everyone's fills visible; prev reads of buf[1-cur] done
        if (kt + 1 < NT) {    // overlap next copy with this tile's WMMAs
            stage128x32(As[1 - cur], Ab + (kt + 1) * BK, K, tid);
            stage128x32(Bs[1 - cur], Bb + (kt + 1) * BK, K, tid);
        }

        const _Float16* as = As[cur];
        const _Float16* bs = Bs[cur];
        Frag a[2], bf[4];
#pragma unroll
        for (int mt = 0; mt < 2; mt++) {
            const _Float16* base = as + (wm * 32 + mt * 16 + laneN) * LDT + half * 8;
            a[mt].q[0] = *(const u32x4*)(base);        // K = half*8 + 0..7
            a[mt].q[1] = *(const u32x4*)(base + 16);   // K = 16 + half*8 + 0..7
        }
#pragma unroll
        for (int nt = 0; nt < 4; nt++) {
            const _Float16* base = bs + (wn * 64 + nt * 16 + laneN) * LDT + half * 8;
            bf[nt].q[0] = *(const u32x4*)(base);       // K = half*8 + 0..7
            bf[nt].q[1] = *(const u32x4*)(base + 16);  // K = 16 + half*8 + 0..7
        }
#pragma unroll
        for (int mt = 0; mt < 2; mt++)
#pragma unroll
            for (int nt = 0; nt < 4; nt++)
                acc[mt][nt] = wmma_f16(a[mt], bf[nt], acc[mt][nt]);
    }

#pragma unroll
    for (int mt = 0; mt < 2; mt++)
#pragma unroll
        for (int nt = 0; nt < 4; nt++) {
            int row0 = m0 + wm * 32 + mt * 16 + half * 8;   // C layout: M = r + 8*half
            int col  = n0 + wn * 64 + nt * 16 + laneN;      //           N = laneN
#pragma unroll
            for (int r = 0; r < 8; r++) {
                float v = acc[mt][nt][r];
                size_t idx = (size_t)(row0 + r) * N + col;
                if (store_half) Ch[idx] = (_Float16)v;
                else            Cf[idx] = v;
            }
        }
}

// ---------------------------------------------------------------- attention
// One block = 128 q rows x one (batch, head); 8 waves x 16 q rows.
// Streaming softmax over key chunks of 32; K tile staged async-b128.
__global__ __launch_bounds__(256) void k_attn(
    const _Float16* __restrict__ Qh,    // [S*B, 1024]   row = s*B + b
    const _Float16* __restrict__ KVh,   // [S*B, 2048]   K cols 0..1023, V cols 1024..2047
    const unsigned char* __restrict__ mask,  // [S, B] nonzero = masked
    _Float16* __restrict__ Oh)          // [S*B, 1024]
{
    __shared__ __align__(16) _Float16 Kt[32 * 72];      // [key][dim], 144B rows
    __shared__ __align__(16) _Float16 Vt[64 * 40];      // [dim][key]
    __shared__ __align__(16) _Float16 Pl[8 * 16 * 40];  // per-wave P scratch [m][j]
    __shared__ float maskf[32];

    const int tid  = threadIdx.x;
    const int wave = tid >> 5, lane = tid & 31;
    const int laneN = lane & 15, half = lane >> 4;
    const int b    = blockIdx.y % BSZ;
    const int head = blockIdx.y / BSZ;
    const int q0   = blockIdx.x * 128 + wave * 16;

    // Q fragments (A layout) kept in registers for the whole key sweep
    Frag qa[2];
    {
        size_t qoff = ((size_t)(q0 + laneN) * BSZ + b) * (N_HEADS * D_HEAD) + head * D_HEAD;
#pragma unroll
        for (int part = 0; part < 2; part++)
#pragma unroll
            for (int v = 0; v < 8; v++) {
                int k = part * 32 + ((v & 3) << 1) + half * 8 + ((v >> 2) << 4);
                qa[part].u[v] = *(const unsigned int*)(Qh + qoff + k);
            }
    }

    float m_r[8], l_r[8];
    v8f zero = {};
    v8f o[4];
#pragma unroll
    for (int r = 0; r < 8; r++) { m_r[r] = -3.0e38f; l_r[r] = 0.0f; }
#pragma unroll
    for (int nt = 0; nt < 4; nt++) o[nt] = zero;

    const int kkey = tid >> 3, kchunk = tid & 7;   // K-tile async mapping

    for (int j0 = 0; j0 < SEQ; j0 += 32) {
        __syncthreads();   // previous iteration's readers are done with Kt/Vt
        // K tile [32 keys][64 dims]: contiguous 128B rows -> async b128
        async_g2l_b128(lds_addr_of(Kt + kkey * 72) + kchunk * 16,
                       KVh + ((size_t)(j0 + kkey) * BSZ + b) * (2 * N_HEADS * D_HEAD)
                           + head * D_HEAD + kchunk * 8);
        // V tile transposed Vt[dim][key]: register-staged 16-bit copy
        _Float16 vtmp[8];
#pragma unroll
        for (int it = 0; it < 8; it++) {
            int e = tid + it * 256, kk = e >> 6, d = e & 63;
            vtmp[it] = KVh[((size_t)(j0 + kk) * BSZ + b) * (2 * N_HEADS * D_HEAD)
                           + N_HEADS * D_HEAD + head * D_HEAD + d];
        }
#pragma unroll
        for (int it = 0; it < 8; it++) {
            int e = tid + it * 256, kk = e >> 6, d = e & 63;
            Vt[d * 40 + kk] = vtmp[it];
        }
        if (tid < 32)
            maskf[tid] = mask[(size_t)(j0 + tid) * BSZ + b] ? -1.0e30f : 0.0f;
        wait_async0();
        __syncthreads();

        if (j0 + 32 < SEQ)   // CDNA5 global_prefetch_b8 of the next K chunk
            __builtin_prefetch(KVh + ((size_t)(j0 + 32) * BSZ + b) * (2 * N_HEADS * D_HEAD)
                               + head * D_HEAD, 0, 3);

        // scores: two 16x16 f32 tiles (keys +0..15 and +16..31), k-dim = 2x32 head dims
        v8f s0 = zero, s1 = zero;
        {
            Frag kb;
            const _Float16* kb0 = Kt + laneN * 72 + half * 8;
            kb.q[0] = *(const u32x4*)(kb0);      kb.q[1] = *(const u32x4*)(kb0 + 16);
            s0 = wmma_f16(qa[0], kb, s0);
            kb.q[0] = *(const u32x4*)(kb0 + 32); kb.q[1] = *(const u32x4*)(kb0 + 48);
            s0 = wmma_f16(qa[1], kb, s0);
            const _Float16* kb1 = Kt + (16 + laneN) * 72 + half * 8;
            kb.q[0] = *(const u32x4*)(kb1);      kb.q[1] = *(const u32x4*)(kb1 + 16);
            s1 = wmma_f16(qa[0], kb, s1);
            kb.q[0] = *(const u32x4*)(kb1 + 32); kb.q[1] = *(const u32x4*)(kb1 + 48);
            s1 = wmma_f16(qa[1], kb, s1);
        }

        // online softmax; row m = r + 8*half lives across the 16 lanes of this half
        const float msk0 = maskf[laneN];
        const float msk1 = maskf[16 + laneN];
        _Float16* pw = Pl + wave * (16 * 40);
#pragma unroll
        for (int r = 0; r < 8; r++) {
            float v0 = s0[r] * QK_SCALE + msk0;
            float v1 = s1[r] * QK_SCALE + msk1;
            float t = fmaxf(v0, v1);
#pragma unroll
            for (int off = 8; off >= 1; off >>= 1)
                t = fmaxf(t, __shfl_xor(t, off, 32));
            float mn    = fmaxf(m_r[r], t);
            float alpha = __expf(m_r[r] - mn);
            float p0 = __expf(v0 - mn);
            float p1 = __expf(v1 - mn);
            float rs = p0 + p1;
#pragma unroll
            for (int off = 8; off >= 1; off >>= 1)
                rs += __shfl_xor(rs, off, 32);
            l_r[r] = l_r[r] * alpha + rs;
            m_r[r] = mn;
#pragma unroll
            for (int nt = 0; nt < 4; nt++) o[nt][r] *= alpha;
            int m = r + half * 8;                    // C layout -> A layout via LDS
            pw[m * 40 + laneN]      = (_Float16)p0;
            pw[m * 40 + 16 + laneN] = (_Float16)p1;
        }
        asm volatile("s_wait_dscnt 0x0" ::: "memory");  // wave-local LDS RAW fence

        Frag pf;  // P as A-matrix 16x32
        {
            const _Float16* pb = pw + laneN * 40 + half * 8;
            pf.q[0] = *(const u32x4*)(pb);
            pf.q[1] = *(const u32x4*)(pb + 16);
        }
#pragma unroll
        for (int nt = 0; nt < 4; nt++) {
            Frag vb;
            const _Float16* vbase = Vt + (nt * 16 + laneN) * 40 + half * 8;
            vb.q[0] = *(const u32x4*)(vbase);
            vb.q[1] = *(const u32x4*)(vbase + 16);
            o[nt] = wmma_f16(pf, vb, o[nt]);
        }
    }

    // normalize and store attn_vec (f16 feeds the Wo GEMM)
#pragma unroll
    for (int nt = 0; nt < 4; nt++) {
        int col = head * D_HEAD + nt * 16 + laneN;
#pragma unroll
        for (int r = 0; r < 8; r++) {
            int s = q0 + r + half * 8;
            float v = o[nt][r] / l_r[r];
            Oh[((size_t)s * BSZ + b) * (N_HEADS * D_HEAD) + col] = (_Float16)v;
        }
    }
}

// ---------------------------------------------------------------- residual + LayerNorm
__global__ __launch_bounds__(256) void k_ln(
    const float* __restrict__ h, const float* __restrict__ ao,
    const float* __restrict__ gamma, const float* __restrict__ beta,
    float* __restrict__ out)
{
    __shared__ float red[256];
    const int row = blockIdx.x;
    const int tid = threadIdx.x;
    const float* hr = h  + (size_t)row * D_MODEL;
    const float* ar = ao + (size_t)row * D_MODEL;
    float x[4];
    float s = 0.f;
#pragma unroll
    for (int i = 0; i < 4; i++) { x[i] = hr[tid + i * 256] + ar[tid + i * 256]; s += x[i]; }
    red[tid] = s; __syncthreads();
    for (int w = 128; w > 0; w >>= 1) { if (tid < w) red[tid] += red[tid + w]; __syncthreads(); }
    float mean = red[0] / D_MODEL;
    __syncthreads();
    float vs = 0.f;
#pragma unroll
    for (int i = 0; i < 4; i++) { float d = x[i] - mean; vs += d * d; }
    red[tid] = vs; __syncthreads();
    for (int w = 128; w > 0; w >>= 1) { if (tid < w) red[tid] += red[tid + w]; __syncthreads(); }
    float rstd = rsqrtf(red[0] / D_MODEL + LN_EPS);
#pragma unroll
    for (int i = 0; i < 4; i++) {
        int c = tid + i * 256;
        out[(size_t)row * D_MODEL + c] = (x[i] - mean) * rstd * gamma[c] + beta[c];
    }
}

// ---------------------------------------------------------------- launch
extern "C" void kernel_launch(void* const* d_in, const int* in_sizes, int n_in,
                              void* d_out, int out_size, void* d_ws, size_t ws_size,
                              hipStream_t stream)
{
    (void)in_sizes; (void)n_in; (void)out_size; (void)ws_size;
    const float*         h     = (const float*)d_in[0];
    const unsigned char* mask  = (const unsigned char*)d_in[1];
    const float*         Wq    = (const float*)d_in[2];
    const float*         Wkv   = (const float*)d_in[3];
    const float*         Wo    = (const float*)d_in[4];
    const float*         gamma = (const float*)d_in[5];
    const float*         beta  = (const float*)d_in[6];
    float*               out   = (float*)d_out;

    const int ROWS = SEQ * BSZ;   // 8192

    char* ws = (char*)d_ws;
    size_t off = 0;
    auto alloc = [&](size_t bytes) {
        char* p = ws + off;
        off += (bytes + 255) & ~(size_t)255;
        return p;
    };
    _Float16* hb    = (_Float16*)alloc((size_t)ROWS * D_MODEL * 2);
    _Float16* WqT   = (_Float16*)alloc((size_t)D_MODEL * D_MODEL * 2);       // [N][K]
    _Float16* WkvT  = (_Float16*)alloc((size_t)2 * D_MODEL * D_MODEL * 2);   // [2048][1024]
    _Float16* WoT   = (_Float16*)alloc((size_t)D_MODEL * D_MODEL * 2);
    _Float16* Qh    = (_Float16*)alloc((size_t)ROWS * D_MODEL * 2);
    _Float16* KVh   = (_Float16*)alloc((size_t)ROWS * 2 * D_MODEL * 2);
    _Float16* AVh   = (_Float16*)alloc((size_t)ROWS * D_MODEL * 2);
    float*    AOf   = (float*)   alloc((size_t)ROWS * D_MODEL * 4);

    k_f32_to_f16<<<ROWS * D_MODEL / 256, 256, 0, stream>>>(h, hb);
    k_w_transpose_f16<<<D_MODEL * D_MODEL / 256, 256, 0, stream>>>(Wq, WqT, D_MODEL, 10);
    k_w_transpose_f16<<<2 * D_MODEL * D_MODEL / 256, 256, 0, stream>>>(Wkv, WkvT, D_MODEL, 11);
    k_w_transpose_f16<<<D_MODEL * D_MODEL / 256, 256, 0, stream>>>(Wo, WoT, D_MODEL, 10);

    dim3 blk(256);
    // Q = h*Wq (8192x1024x1024), f16 out
    k_gemm_f16<<<dim3(D_MODEL / BN, ROWS / BM), blk, 0, stream>>>(
        hb, WqT, nullptr, Qh, ROWS, D_MODEL, D_MODEL, 1);
    // KV = h*Wkv (8192x2048x1024), f16 out
    k_gemm_f16<<<dim3(2 * D_MODEL / BN, ROWS / BM), blk, 0, stream>>>(
        hb, WkvT, nullptr, KVh, ROWS, 2 * D_MODEL, D_MODEL, 1);
    // flash attention: 16 q-blocks x (4 batch * 16 heads)
    k_attn<<<dim3(SEQ / 128, BSZ * N_HEADS), blk, 0, stream>>>(Qh, KVh, mask, AVh);
    // attn_out = attn_vec * Wo, f32 out
    k_gemm_f16<<<dim3(D_MODEL / BN, ROWS / BM), blk, 0, stream>>>(
        AVh, WoT, AOf, nullptr, ROWS, D_MODEL, D_MODEL, 0);
    // out = LayerNorm(h + attn_out)
    k_ln<<<ROWS, blk, 0, stream>>>(h, AOf, gamma, beta, out);
}
